// CapsuleLayer_77257871720720
// MI455X (gfx1250) — matrix operand order repaired and checked by reading, not compiled, not run on hardware
//
#include <hip/hip_runtime.h>

typedef __attribute__((ext_vector_type(16))) _Float16 v16h;
typedef __attribute__((ext_vector_type(8)))  float    v8f;

#define B_  16
#define P_  1024
#define J_  32
#define DP_ 64
#define D_  64

// ---------------------------------------------------------------------------
// Kernel 1: u[b,p,j,d] = sum_e inp[b,p,e] * W[p,j,e,d] + bias[p,j,d]
// One wave32 per (p,j): C[16x64] = A[16x64] x B[64x64] using 8x
// v_wmma_f32_16x16x32_f16 (4 N-tiles x 2 K-tiles), f32 accumulate.
// ---------------------------------------------------------------------------
__global__ __launch_bounds__(256) void caps_u_wmma(
    const float* __restrict__ inp,   // [B,P,DP]
    const float* __restrict__ W,     // [P,J,DP,D]
    const float* __restrict__ bias,  // [P,J,D]
    float* __restrict__ u)           // [B,P,J,D]
{
    const int lane = threadIdx.x & 31;
    const int wid  = blockIdx.x * (blockDim.x >> 5) + (threadIdx.x >> 5);
    const int p = wid >> 5;   // / J_
    const int j = wid & 31;   // % J_
    if (p >= P_) return;

    const int  m  = lane & 15;            // A row (=batch), B column-local
    const bool hi = (lane & 16) != 0;     // lane half

    v8f acc[4] = {v8f{}, v8f{}, v8f{}, v8f{}};

    const float* arow = inp + (size_t)m * (P_ * DP_) + (size_t)p * DP_;
    const float* wt   = W   + ((size_t)p * J_ + j) * (DP_ * D_);

    #pragma unroll
    for (int kt = 0; kt < DP_; kt += 32) {
        // A fragment 16x32 f16 (ISA 7.12.2 16-bit A layout):
        // lane<16: K = kb..kb+7 (v0-3), kb+16..kb+23 (v4-7), kb=0
        // lane>=16: same with kb=8
        const int kba = hi ? 8 : 0;
        v16h a;
        #pragma unroll
        for (int i = 0; i < 8; ++i)
            a[i]     = (_Float16)arow[kt + kba + i];
        #pragma unroll
        for (int i = 0; i < 8; ++i)
            a[8 + i] = (_Float16)arow[kt + kba + 16 + i];

        // B fragments 32x16 f16: lane = column, same K order as A:
        // K = kbb+i (i<8), kbb+16+(i-8) (i>=8), kbb = 0 / 8 per lane half.
        const int kbb = hi ? 8 : 0;
        #pragma unroll
        for (int nt = 0; nt < 4; ++nt) {
            const int d = nt * 16 + m;
            v16h bf;
            #pragma unroll
            for (int i = 0; i < 16; ++i)
                bf[i] = (_Float16)wt[(size_t)(kt + kbb + i + (i & 8)) * D_ + d];
            acc[nt] = __builtin_amdgcn_wmma_f32_16x16x32_f16(
                false, a, false, bf, (short)0, acc[nt], false, false);
        }
    }

    // C/D layout: VGPR r holds M=r (lanes 0-15) / M=r+8 (lanes 16-31), N=lane&15
    const float* bp = bias + ((size_t)p * J_ + j) * D_;
    #pragma unroll
    for (int nt = 0; nt < 4; ++nt) {
        const int d = nt * 16 + m;
        const float bv = bp[d];
        #pragma unroll
        for (int r = 0; r < 8; ++r) {
            const int bb = r + (hi ? 8 : 0);
            u[(((size_t)bb * P_ + p) * J_ + j) * D_ + d] = acc[nt][r] + bv;
        }
    }
}

// ---------------------------------------------------------------------------
// Zero-fill (d_ws is poisoned; bij must start at 0)
// ---------------------------------------------------------------------------
__global__ void caps_zero(float* __restrict__ ptr, int n) {
    int i = blockIdx.x * blockDim.x + threadIdx.x;
    if (i < n) ptr[i] = 0.0f;
}

// ---------------------------------------------------------------------------
// Softmax over J: one wave32 per (b,p), lane == j. Pure shuffle reductions.
// ---------------------------------------------------------------------------
__global__ __launch_bounds__(256) void caps_softmax(
    const float* __restrict__ bij, float* __restrict__ cij)
{
    const int lane = threadIdx.x & 31;
    const int w = blockIdx.x * (blockDim.x >> 5) + (threadIdx.x >> 5);
    if (w >= B_ * P_) return;
    const size_t idx = (size_t)w * J_ + lane;
    const float x = bij[idx];
    float mx = x;
    #pragma unroll
    for (int off = 16; off >= 1; off >>= 1) mx = fmaxf(mx, __shfl_xor(mx, off, 32));
    const float e = __expf(x - mx);
    float s = e;
    #pragma unroll
    for (int off = 16; off >= 1; off >>= 1) s += __shfl_xor(s, off, 32);
    cij[idx] = e / s;
}

// ---------------------------------------------------------------------------
// Fused per-(b,j) block: s = sum_p c*u ; v = squash(s) ;
// if !last: bij += <v, u> per p ; if last: write v to out.
// ---------------------------------------------------------------------------
__global__ __launch_bounds__(256) void caps_route(
    const float* __restrict__ u, const float* __restrict__ cij,
    float* __restrict__ bij, float* __restrict__ out, int last)
{
    __shared__ float red[256];
    __shared__ float sv[D_];
    __shared__ float vv[D_];

    const int b = blockIdx.x >> 5;   // / J_
    const int j = blockIdx.x & 31;   // % J_
    const int t = threadIdx.x;
    const int d  = t & 63;
    const int pg = t >> 6;           // 4 p-groups

    // Phase 1: weighted sum over P (coalesced 256B rows of u)
    float acc = 0.0f;
    for (int p = pg; p < P_; p += 4) {
        const size_t bpj = ((size_t)b * P_ + p) * J_ + j;
        acc = fmaf(cij[bpj], u[bpj * D_ + d], acc);
    }
    red[t] = acc;
    __syncthreads();
    if (pg == 0)
        sv[d] = red[d] + red[d + 64] + red[d + 128] + red[d + 192];
    __syncthreads();

    // squash: scale = sqrt(s2)/(1+s2)  (== (s2/(1+s2))/sqrt(s2))
    float s2 = 0.0f;
    #pragma unroll 8
    for (int i = 0; i < D_; ++i) { const float x = sv[i]; s2 = fmaf(x, x, s2); }
    const float scale = sqrtf(s2) / (1.0f + s2);

    if (pg == 0) {
        const float v = sv[d] * scale;
        vv[d] = v;
        if (last) out[((size_t)b * J_ + j) * D_ + d] = v;
    }
    __syncthreads();

    // Phase 2: agreement bij[b,p,j] += dot(v, u[b,p,j,:]) (u is L2-resident)
    if (!last) {
        const int w = t >> 5, lane = t & 31;
        for (int p = w; p < P_; p += 8) {
            const size_t base = (((size_t)b * P_ + p) * J_ + j) * D_;
            float a = fmaf(vv[lane], u[base + lane],
                           vv[lane + 32] * u[base + lane + 32]);
            #pragma unroll
            for (int off = 16; off >= 1; off >>= 1) a += __shfl_xor(a, off, 32);
            if (lane == 0) bij[((size_t)b * P_ + p) * J_ + j] += a;
        }
    }
}

// ---------------------------------------------------------------------------
// Host launcher. Workspace layout (f32):
//   u   : B*P*J*D = 33,554,432 floats (134.2 MB)
//   bij : B*P*J   =    524,288 floats (2 MB)
//   cij : B*P*J   =    524,288 floats (2 MB)
// total ~138.4 MB
// ---------------------------------------------------------------------------
extern "C" void kernel_launch(void* const* d_in, const int* in_sizes, int n_in,
                              void* d_out, int out_size, void* d_ws, size_t ws_size,
                              hipStream_t stream) {
    const float* inp  = (const float*)d_in[0];   // [16,1024,64]
    const float* W    = (const float*)d_in[1];   // [1,1024,32,64,64]
    const float* bias = (const float*)d_in[2];   // [1,1024,32,64]
    float* out = (float*)d_out;                  // [16,32,64]

    float* u   = (float*)d_ws;
    float* bij = u   + (size_t)B_ * P_ * J_ * D_;
    float* cij = bij + (size_t)B_ * P_ * J_;

    // 32768 waves (one per (p,j)), 8 waves/block
    caps_u_wmma<<<(P_ * J_) / 8, 256, 0, stream>>>(inp, W, bias, u);

    const int nb = B_ * P_ * J_;
    caps_zero<<<(nb + 255) / 256, 256, 0, stream>>>(bij, nb);

    for (int it = 0; it < 3; ++it) {
        caps_softmax<<<(B_ * P_) / 8, 256, 0, stream>>>(bij, cij);
        caps_route<<<B_ * J_, 256, 0, stream>>>(u, cij, bij, out, (it == 2) ? 1 : 0);
    }
}
